// Attention_12945031430357
// MI455X (gfx1250) — matrix operand order repaired and checked by reading, not compiled, not run on hardware
//
#include <hip/hip_runtime.h>
#include <math.h>

// ---------------------------------------------------------------------------
// Attention: qk = x@W^T + b ; dots = K Q^T / sqrt(D) ; softmax ; out = P @ x
// B=8, N=2048, D=1024. Compute-bound (~206 GFLOP vs ~136 MiB I/O) =>
// v_wmma_f32_16x16x32_f16 everywhere, LDS-tiled GEMM (64 FLOP/B global AI),
// async global->LDS staging (ASYNCcnt path) for operand reuse.
// ---------------------------------------------------------------------------

typedef __attribute__((ext_vector_type(16))) _Float16 v16h;
typedef __attribute__((ext_vector_type(8)))  float    v8f;

#define B_ 8
#define N_ 2048
#define D_ 1024
#define SCALE_ 0.03125f   // 1/sqrt(1024)

__device__ __forceinline__ v8f wmma_f16(v16h a, v16h b, v8f c) {
  return __builtin_amdgcn_wmma_f32_16x16x32_f16(false, a, false, b,
                                                (short)0, c, false, false);
}

// Low 32 bits of a shared-aperture flat address == wave-relative LDS offset.
__device__ __forceinline__ unsigned lds_off(const void* p) {
  return (unsigned)(unsigned long long)p;
}
// Async 16B copy global -> LDS (per-lane addresses, tracked by ASYNCcnt).
__device__ __forceinline__ void async_cp16(unsigned ldsByteOff, const void* gsrc) {
  asm volatile("global_load_async_to_lds_b128 %0, %1, off"
               :: "v"(ldsByteOff), "v"(gsrc) : "memory");
}
__device__ __forceinline__ void wait_async0() {
  asm volatile("s_wait_asynccnt 0" ::: "memory");
}

// Fragment layouts used here (wave32, 16-bit data; contiguous K, not the ISA table order):
//  A (16Mx32K): lane L: row M=L&15; half h -> K = h + 16*(L>>4)
//  B (32Kx16N): lane L: col N=L&15; half h -> K = h + 16*(L>>4)
//  C (16x16 f32): vgpr v -> row M = v + 8*(L>>4), col N = L&15

// ---- pack x into A-fragment order (tiles: rt in [0,1024), ds in [0,32)) ----
__global__ __launch_bounds__(256) void pack_xA(const float* __restrict__ x,
                                               _Float16* __restrict__ dst) {
  int gid = blockIdx.x * 256 + threadIdx.x;
  int t = gid >> 5, lane = gid & 31;
  int rt = t >> 5, ds = t & 31;
  int M = lane & 15, g = lane >> 4;
  const float* src = x + (size_t)(rt * 16 + M) * D_ + ds * 32;
  v16h frag;
#pragma unroll
  for (int h = 0; h < 16; ++h) {
    int K = h + (g << 4);
    frag[h] = (_Float16)src[K];
  }
  *(v16h*)(dst + ((size_t)t << 9) + (lane << 4)) = frag;
}

// ---- pack W into B-fragment order (tiles: ot in [0,128), ds in [0,32)) -----
__global__ __launch_bounds__(256) void pack_wB(const float* __restrict__ W,
                                               _Float16* __restrict__ dst) {
  int gid = blockIdx.x * 256 + threadIdx.x;
  int t = gid >> 5, lane = gid & 31;
  int ot = t >> 5, ds = t & 31;
  int Nc = lane & 15, g = lane >> 4;
  const float* src = W + (size_t)(ot * 16 + Nc) * D_ + ds * 32 + g * 16;
  v16h frag;
#pragma unroll
  for (int h = 0; h < 16; ++h) frag[h] = (_Float16)src[h];
  *(v16h*)(dst + ((size_t)t << 9) + (lane << 4)) = frag;
}

// ---- pack x into B-fragment order for P@x (tiles: (b, js, dt)) -------------
__global__ __launch_bounds__(256) void pack_xB(const float* __restrict__ x,
                                               _Float16* __restrict__ dst) {
  int gid = blockIdx.x * 256 + threadIdx.x;
  int t = gid >> 5, lane = gid & 31;
  int b = t >> 12, rem = t & 4095;
  int js = rem >> 6, dt = rem & 63;
  int Nc = lane & 15, g = lane >> 4;
  const float* src = x + ((size_t)b * N_ + js * 32 + g * 16) * D_ + dt * 16 + Nc;
  v16h frag;
#pragma unroll
  for (int h = 0; h < 16; ++h) frag[h] = (_Float16)src[(size_t)h * D_];
  *(v16h*)(dst + ((size_t)t << 9) + (lane << 4)) = frag;
}

// ---- qk = x @ W^T + b : 128x128 macro-tile / workgroup, LDS double-buffer --
// Per K-chunk (32): stage 8 A-tiles + 8 B-tiles (16 KB) via async->LDS.
// Wave w owns row subtile w (16 rows) x 8 col subtiles: 8 WMMA / chunk.
__global__ __launch_bounds__(256) void qk_gemm(const _Float16* __restrict__ xA,
                                               const _Float16* __restrict__ wB,
                                               const float* __restrict__ bias,
                                               _Float16* __restrict__ qB,
                                               _Float16* __restrict__ kA) {
  __shared__ __align__(32) _Float16 stage[2][16 * 512];   // 2 x 16 tiles
  int w = threadIdx.x >> 5, lane = threadIdx.x & 31;
  int R  = blockIdx.x >> 4;      // row macro  [0,128)
  int Cm = blockIdx.x & 15;      // col macro  [0,16)
  int rt0 = R * 8, ot0 = Cm * 8;

  const char* xAb = (const char*)xA;
  const char* wBb = (const char*)wB;

  // issue 16 KB chunk `ds` into stage[p]: 256 threads x 4 x 16B
  auto issue_chunk = [&](int p, int ds) {
    unsigned base = lds_off(&stage[p][0]);
#pragma unroll
    for (int q = 0; q < 4; ++q) {
      int off = q * 4096 + threadIdx.x * 16;           // [0, 16384)
      int tI = off >> 10, wo = off & 1023;             // tile, byte-in-tile
      const char* src = (tI < 8)
          ? xAb + ((((size_t)(rt0 + tI)) * 32 + ds) << 10) + wo
          : wBb + ((((size_t)(ot0 + tI - 8)) * 32 + ds) << 10) + wo;
      async_cp16(base + off, src);
    }
  };

  v8f acc[8] = {};
  issue_chunk(0, 0);
  for (int ds = 0; ds < 32; ++ds) {
    int p = ds & 1;
    wait_async0();
    __syncthreads();                       // chunk ds visible; prev compute done
    if (ds + 1 < 32) issue_chunk(p ^ 1, ds + 1);
    const _Float16* As = &stage[p][(size_t)w << 9];
    const _Float16* Bs = &stage[p][(size_t)8 << 9];
    // preload A + all 8 B fragments, then burst 8 independent WMMA chains
    v16h a = *(const v16h*)(As + (lane << 4));
    v16h bw[8];
#pragma unroll
    for (int cb = 0; cb < 8; ++cb)
      bw[cb] = *(const v16h*)(Bs + ((size_t)cb << 9) + (lane << 4));
#pragma unroll
    for (int cb = 0; cb < 8; ++cb)
      acc[cb] = wmma_f16(a, bw[cb], acc[cb]);
    __syncthreads();                       // done reading stage[p] before reuse
  }

  // epilogue: +bias, f16, scatter q (B-frag) / k (A-frag)
  int Nc = lane & 15, rowg = (lane >> 4) << 3;
  int rbase = (rt0 + w) * 16;
#pragma unroll
  for (int cb = 0; cb < 8; ++cb) {
    int o = (ot0 + cb) * 16 + Nc;
    float bv = bias[o];
#pragma unroll
    for (int v = 0; v < 8; ++v) {
      int r = rbase + v + rowg;            // flattened token
      int bb = r >> 11, ij = r & 2047;
      _Float16 hv = (_Float16)(acc[cb][v] + bv);
      if (o < D_) {                        // q -> B-fragment (b, d/32, j/16)
        int K = o & 31;
        size_t tile = ((size_t)(bb * 32 + (o >> 5))) * 128 + (ij >> 4);
        qB[(tile << 9) + ((ij & 15) + ((K >> 4) << 4)) * 16 + (K & 15)] = hv;
      } else {                             // k -> A-fragment (b, i/16, d/32)
        int d = o - D_;
        int K = d & 31;
        size_t tile = ((size_t)(bb * 128 + (ij >> 4))) * 32 + (d >> 5);
        int lp = (ij & 15) + ((K >> 4) << 4);
        int hp = K & 15;
        kA[(tile << 9) + lp * 16 + hp] = hv;
      }
    }
  }
}

// ---- fused flash attention ------------------------------------------------
// 1 block = (batch b, 16 rows i). K-fragments (32 KB) staged once via async.
// 8 waves: wave w owns output cols [128w, 128w+128).
__global__ __launch_bounds__(256) void flash_attn(const _Float16* __restrict__ kA,
                                                  const _Float16* __restrict__ qB,
                                                  const _Float16* __restrict__ xB,
                                                  float* __restrict__ out) {
  __shared__ __align__(32) _Float16 Kl[32 * 512];        // 32 KB: K A-fragments
  __shared__ float Slds[16 * 128];
  __shared__ __align__(32) _Float16 Plds[4 * 512];
  __shared__ float mS[16], lS[16], aS[16];
  int w = threadIdx.x >> 5, lane = threadIdx.x & 31;
  int b = blockIdx.x >> 7;
  int it = blockIdx.x & 127;
  int rowg = (lane >> 4) << 3;
  int colN = lane & 15;

  // stage all 32 K fragment tiles for (b, it): 256 threads x 8 x 16B
  {
    const char* src = (const char*)kA + (((size_t)b * 128 + it) * 32 << 10);
    unsigned base = lds_off(&Kl[0]);
#pragma unroll
    for (int rr = 0; rr < 8; ++rr) {
      int off = rr * 4096 + threadIdx.x * 16;
      async_cp16(base + off, src + off);
    }
  }
  if (threadIdx.x < 16) { mS[threadIdx.x] = -3.0e38f; lS[threadIdx.x] = 0.f; }
  v8f acc[8] = {};
  wait_async0();
  __syncthreads();

  for (int jb = 0; jb < 16; ++jb) {
    // ---- phase 2: wave w computes S subtile at j-tile jb*8+w (K from LDS)
    // two interleaved accumulator chains for matrix-pipe ILP
    int jt = jb * 8 + w;
    v8f s0 = {}, s1 = {};
#pragma unroll 4
    for (int ds = 0; ds < 32; ds += 2) {
      v16h a0  = *(const v16h*)(Kl + ((size_t)ds << 9) + (lane << 4));
      v16h a1  = *(const v16h*)(Kl + ((size_t)(ds + 1) << 9) + (lane << 4));
      v16h qb0 = *(const v16h*)(qB + ((((size_t)b * 32 + ds) * 128 + jt) << 9) + (lane << 4));
      v16h qb1 = *(const v16h*)(qB + ((((size_t)b * 32 + ds + 1) * 128 + jt) << 9) + (lane << 4));
      s0 = wmma_f16(a0, qb0, s0);
      s1 = wmma_f16(a1, qb1, s1);
    }
#pragma unroll
    for (int v = 0; v < 8; ++v)
      Slds[(v + rowg) * 128 + w * 16 + colN] = (s0[v] + s1[v]) * SCALE_;
    __syncthreads();

    // ---- online softmax: wave w owns rows 2w, 2w+1; 8 cols per lane
    {
      int row = 2 * w + (lane >> 4);
      int c0 = (lane & 15) * 8;
      const float* Sr = &Slds[row * 128 + c0];
      float bm = -3.0e38f;
#pragma unroll
      for (int cI = 0; cI < 8; ++cI) bm = fmaxf(bm, Sr[cI]);
#pragma unroll
      for (int m = 1; m <= 8; m <<= 1) bm = fmaxf(bm, __shfl_xor(bm, m, 32));
      float mprev = mS[row];
      float mnew  = fmaxf(mprev, bm);
      float alpha = __expf(mprev - mnew);
      float psum = 0.f;
#pragma unroll
      for (int cI = 0; cI < 8; ++cI) {
        int cc = c0 + cI;
        float p = __expf(Sr[cI] - mnew);
        psum += p;
        int K = cc & 31;                   // P -> A-fragment order in LDS
        int lp = row + ((K >> 4) << 4);
        int hp = K & 15;
        Plds[((cc >> 5) << 9) + lp * 16 + hp] = (_Float16)p;
      }
#pragma unroll
      for (int m = 1; m <= 8; m <<= 1) psum += __shfl_xor(psum, m, 32);
      if ((lane & 15) == 0) {
        mS[row] = mnew;
        lS[row] = lS[row] * alpha + psum;
        aS[row] = alpha;
      }
    }
    __syncthreads();

    // ---- rescale accumulators by alpha[row]
#pragma unroll
    for (int v = 0; v < 8; ++v) {
      float al = aS[v + rowg];
#pragma unroll
      for (int ct = 0; ct < 8; ++ct) acc[ct][v] *= al;
    }

    // ---- phase 3: acc[ct] += P(16x32) @ x(32x16) over the 128-wide j block
#pragma unroll
    for (int jtl = 0; jtl < 4; ++jtl) {
      v16h pa = *(const v16h*)(Plds + ((size_t)jtl << 9) + (lane << 4));
      size_t jsBase = ((size_t)b * 64 + jb * 4 + jtl) * 64 + w * 8;
#pragma unroll
      for (int ct = 0; ct < 8; ++ct) {
        v16h xb = *(const v16h*)(xB + ((jsBase + ct) << 9) + (lane << 4));
        acc[ct] = wmma_f16(pa, xb, acc[ct]);
      }
    }
    __syncthreads();
  }

  // ---- epilogue: out = acc / l
#pragma unroll
  for (int v = 0; v < 8; ++v) {
    int row = v + rowg;
    float inv = 1.0f / lS[row];
    size_t base = ((size_t)b * N_ + it * 16 + row) * D_ + w * 128 + colN;
#pragma unroll
    for (int ct = 0; ct < 8; ++ct) out[base + ct * 16] = acc[ct][v] * inv;
  }
}

// ---------------------------------------------------------------------------
extern "C" void kernel_launch(void* const* d_in, const int* in_sizes, int n_in,
                              void* d_out, int out_size, void* d_ws, size_t ws_size,
                              hipStream_t stream) {
  const float* x    = (const float*)d_in[0];   // (8, 2048, 1024) f32
  const float* W    = (const float*)d_in[1];   // (2048, 1024)    f32
  const float* bias = (const float*)d_in[2];   // (2048,)         f32
  float* out = (float*)d_out;                  // (8, 2048, 1024) f32

  // workspace: fragment-ordered f16 buffers (tile = 512 halves = 1 KiB)
  _Float16* xA = (_Float16*)d_ws;              // 32768 tiles (32 MiB)
  _Float16* wB = xA + (size_t)32768 * 512;     //  4096 tiles ( 4 MiB)
  _Float16* xB = wB + (size_t)4096 * 512;      // 32768 tiles (32 MiB)
  _Float16* qB = xB + (size_t)32768 * 512;     // 32768 tiles (32 MiB)
  _Float16* kA = qB + (size_t)32768 * 512;     // 32768 tiles (32 MiB)

  pack_xA<<<4096, 256, 0, stream>>>(x, xA);
  pack_wB<<<512, 256, 0, stream>>>(W, wB);
  pack_xB<<<4096, 256, 0, stream>>>(x, xB);
  qk_gemm<<<2048, 256, 0, stream>>>(xA, wB, bias, qB, kA);
  flash_attn<<<1024, 256, 0, stream>>>(kA, qB, xB, out);
}
